// Attention_22514218565864
// MI455X (gfx1250) — matrix operand order repaired and checked by reading, not compiled, not run on hardware
//
#include <hip/hip_runtime.h>

// Fused linear-attention (no softmax => (QK^T)V == Q(K^T V)).
//   Kernel 0: convert x, Wq, Wk, Wv to f16 (one shot; keeps GEMM loops pure f16)
//   Kernel 1: Q = xWq^T+bq (f16, row-major), Kt/Vt = (xW^T+b)^T (f16, [B][D][S])
//   Kernel 2: P[b][d2][d1] = sum_t V[t][d2]*K[t][d1]  -- async-LDS double-buffered
//   Kernel 3: out[t][d2] = scale * sum_d1 P[d2][d1]*Q[t][d1] (contiguous stores)
// All GEMMs use v_wmma_f32_16x16x32_f16 (wave32), f32 accumulate.

typedef _Float16 f16;
typedef __attribute__((ext_vector_type(16))) _Float16 v16h;
typedef __attribute__((ext_vector_type(8)))  _Float16 v8h;
typedef __attribute__((ext_vector_type(8)))  float    v8f;
typedef __attribute__((ext_vector_type(4)))  float    v4f;
typedef __attribute__((ext_vector_type(4)))  int      v4i;

#ifndef __has_builtin
#define __has_builtin(x) 0
#endif

#if __has_builtin(__builtin_amdgcn_global_load_async_to_lds_b128)
#define HAVE_ASYNC_LDS 1
#else
#define HAVE_ASYNC_LDS 0
#endif

#define DEV __device__ __forceinline__

DEV v8f wmma_f16(v16h a, v16h b, v8f c) {
  return __builtin_amdgcn_wmma_f32_16x16x32_f16(false, a, false, b, (short)0, c,
                                                false, false);
}

DEV v16h cat8(v8h lo, v8h hi) {
  return __builtin_shufflevector(lo, hi, 0, 1, 2, 3, 4, 5, 6, 7, 8, 9, 10, 11,
                                 12, 13, 14, 15);
}

// Copy 16B global -> LDS. Async (ASYNCcnt-tracked) when the builtin exists.
// Toolchain signature (from diagnostics): (int4 __device__*, int4 __shared__*,
// int imm_offset, int cpol)
DEV void copy16_g2l(const f16* g, f16* l) {
#if HAVE_ASYNC_LDS
  typedef __attribute__((address_space(1))) v4i gv4i;
  typedef __attribute__((address_space(3))) v4i lv4i;
  __builtin_amdgcn_global_load_async_to_lds_b128((gv4i*)(f16*)g, (lv4i*)l, 0, 0);
#else
  *(v8h*)l = *(const v8h*)g;
#endif
}

DEV void wait_async_all() {
#if HAVE_ASYNC_LDS
#if __has_builtin(__builtin_amdgcn_s_wait_asynccnt)
  __builtin_amdgcn_s_wait_asynccnt(0);
#else
  asm volatile("s_wait_asynccnt 0x0" ::: "memory");
#endif
#endif
}

// A fragment (16x32, f16) for tile at (m0,k0); src row-major f16, leading dim ld.
DEV v16h load_a_f16(const f16* __restrict__ p, int ld, int m0, int k0, int lane) {
  const int lrow = lane & 15, half = lane >> 4;
  const f16* r = p + (size_t)(m0 + lrow) * ld + (k0 + half * 8);
  v8h lo = *(const v8h*)(r);
  v8h hi = *(const v8h*)(r + 16);
  return cat8(lo, hi);
}

// B fragment (32x16) for tile at (n0,k0); src row-major [n][k], leading dim ld.
DEV v16h load_b_f16(const f16* __restrict__ p, int ld, int n0, int k0, int lane) {
  const int col = lane & 15, kh = lane >> 4;
  const f16* r = p + (size_t)(n0 + col) * ld + (k0 + kh * 8);
  v8h lo = *(const v8h*)(r);
  v8h hi = *(const v8h*)(r + 16);
  return cat8(lo, hi);
}

// ---------------------------------------------------------------------------
// Kernel 0: f32 -> f16 elementwise convert, 8 elems/thread, contiguous b128.
// ---------------------------------------------------------------------------
__global__ __launch_bounds__(256) void cvt_f16_kernel(
    const float* __restrict__ in, f16* __restrict__ out) {
  const size_t i = ((size_t)blockIdx.x * blockDim.x + threadIdx.x) * 8;
  v4f a0 = *(const v4f*)(in + i);
  v4f a1 = *(const v4f*)(in + i + 4);
  v8h o;
#pragma unroll
  for (int j = 0; j < 4; ++j) {
    o[j]     = (f16)a0[j];
    o[4 + j] = (f16)a1[j];
  }
  *(v8h*)(out + i) = o;
}

// ---------------------------------------------------------------------------
// Kernel 1: QKV projection. grid = (M/128, D/64), block = 256 (8 waves, 4x2).
// ---------------------------------------------------------------------------
__global__ __launch_bounds__(256) void qkv_proj_kernel(
    const f16* __restrict__ xh,
    const f16* __restrict__ Wqh, const float* __restrict__ bq,
    const f16* __restrict__ Wkh, const float* __restrict__ bk,
    const f16* __restrict__ Wvh, const float* __restrict__ bv,
    f16* __restrict__ Qh, f16* __restrict__ Kt, f16* __restrict__ Vt) {
  constexpr int D = 768, S = 4096;
  const int lane = threadIdx.x & 31;
  const int wave = threadIdx.x >> 5;
  const int wm = wave & 3, wn = wave >> 2;
  const int blockM = blockIdx.x * 128;
  const int m0 = blockM + wm * 32;
  const int n0 = blockIdx.y * 64 + wn * 32;

  v8f aQ[2][2] = {};
  v8f aK[2][2] = {};
  v8f aV[2][2] = {};

  for (int k0 = 0; k0 < D; k0 += 32) {
    v16h a0 = load_a_f16(xh, D, m0, k0, lane);
    v16h a1 = load_a_f16(xh, D, m0 + 16, k0, lane);
    {
      v16h b0 = load_b_f16(Wqh, D, n0, k0, lane);
      v16h b1 = load_b_f16(Wqh, D, n0 + 16, k0, lane);
      aQ[0][0] = wmma_f16(a0, b0, aQ[0][0]);
      aQ[0][1] = wmma_f16(a0, b1, aQ[0][1]);
      aQ[1][0] = wmma_f16(a1, b0, aQ[1][0]);
      aQ[1][1] = wmma_f16(a1, b1, aQ[1][1]);
    }
    {
      v16h b0 = load_b_f16(Wkh, D, n0, k0, lane);
      v16h b1 = load_b_f16(Wkh, D, n0 + 16, k0, lane);
      aK[0][0] = wmma_f16(a0, b0, aK[0][0]);
      aK[0][1] = wmma_f16(a0, b1, aK[0][1]);
      aK[1][0] = wmma_f16(a1, b0, aK[1][0]);
      aK[1][1] = wmma_f16(a1, b1, aK[1][1]);
    }
    {
      v16h b0 = load_b_f16(Wvh, D, n0, k0, lane);
      v16h b1 = load_b_f16(Wvh, D, n0 + 16, k0, lane);
      aV[0][0] = wmma_f16(a0, b0, aV[0][0]);
      aV[0][1] = wmma_f16(a0, b1, aV[0][1]);
      aV[1][0] = wmma_f16(a1, b0, aV[1][0]);
      aV[1][1] = wmma_f16(a1, b1, aV[1][1]);
    }
  }

  const int col = lane & 15, hb = lane >> 4;
  const int b = blockM >> 12;  // 4096 rows per batch; 128-row tiles never straddle
#pragma unroll
  for (int i = 0; i < 2; ++i) {
#pragma unroll
    for (int j = 0; j < 2; ++j) {
      const int mb = m0 + i * 16 + hb * 8;  // 8 consecutive global rows
      const int nb = n0 + j * 16 + col;     // output feature (fixed per lane)
      const float biasq = bq[nb];
#pragma unroll
      for (int r = 0; r < 8; ++r)
        Qh[(size_t)(mb + r) * D + nb] = (f16)(aQ[i][j][r] + biasq);

      const float biask = bk[nb];
      const float biasv = bv[nb];
      const int t = mb & (S - 1);
      v8h pk, pv;
#pragma unroll
      for (int r = 0; r < 8; ++r) {
        pk[r] = (f16)(aK[i][j][r] + biask);
        pv[r] = (f16)(aV[i][j][r] + biasv);
      }
      // transposed store: one contiguous 16B store per lane
      *(v8h*)(Kt + ((size_t)b * D + nb) * S + t) = pk;
      *(v8h*)(Vt + ((size_t)b * D + nb) * S + t) = pv;
    }
  }
}

// ---------------------------------------------------------------------------
// Kernel 2: P[b][d2][d1] = sum_t K[t][d1]*V[t][d2].  A = Kt rows, B = Vt rows.
// Async-LDS double-buffered: A tile 128x32, B tile 64x32, rows padded to 40 f16
// (80B: keeps ds_load_b128 16B-aligned; 20-bank stride => conflict-free groups).
// grid = (D/128, D/64, B), block = 256 (8 waves, 4x2).
// ---------------------------------------------------------------------------
__global__ __launch_bounds__(256) void kv_outer_kernel(
    const f16* __restrict__ Kt, const f16* __restrict__ Vt,
    f16* __restrict__ P) {
  constexpr int D = 768, S = 4096, PAD = 40, NITER = S / 32;
  __shared__ f16 sA[2][128][PAD];
  __shared__ f16 sB[2][64][PAD];

  const int tid = threadIdx.x;
  const int lane = tid & 31;
  const int wave = tid >> 5;
  const int wm = wave & 3, wn = wave >> 2;
  const int mBlk = blockIdx.x * 128;  // d1
  const int nBlk = blockIdx.y * 64;   // d2
  const int b = blockIdx.z;
  const f16* A = Kt + (size_t)b * D * S;
  const f16* Bp = Vt + (size_t)b * D * S;

  // Per-thread copy slots: A = 4 16B-chunks/thread (ar0 slots alone cover all 512), B = 256 (1/thread)
  const int ar0 = tid >> 1, ao0 = (tid & 1) * 8;           // rows 0..127: chunks 0..511
  const int ar1 = (tid + 256) >> 1, ao1 = ao0;             // rows 128..255, past the 128-row tile
  const int br = tid >> 2, bo = (tid & 3) * 8;

  auto stage = [&](int buf, int k0) {
    copy16_g2l(A + (size_t)(mBlk + ar0) * S + k0 + ao0 + 0,  &sA[buf][ar0][ao0]);
    copy16_g2l(A + (size_t)(mBlk + ar0) * S + k0 + ao0 + 16, &sA[buf][ar0][ao0 + 16]);
    copy16_g2l(A + (size_t)(mBlk + ar1) * S + k0 + ao1 + 0,  &sA[buf][ar1][ao1]);
    copy16_g2l(A + (size_t)(mBlk + ar1) * S + k0 + ao1 + 16, &sA[buf][ar1][ao1 + 16]);
    copy16_g2l(Bp + (size_t)(nBlk + br) * S + k0 + bo, &sB[buf][br][bo]);
  };

  v8f acc[2][2] = {};
  const int lrow = lane & 15, half = lane >> 4;

  stage(0, 0);
  for (int kt = 0; kt < NITER; ++kt) {
    const int buf = kt & 1;
    wait_async_all();
    __syncthreads();
    if (kt + 1 < NITER) stage(buf ^ 1, (kt + 1) * 32);

    const f16* pa0 = &sA[buf][wm * 32 + lrow][half * 8];
    const f16* pa1 = &sA[buf][wm * 32 + 16 + lrow][half * 8];
    const f16* pb0 = &sB[buf][wn * 32 + lrow][half * 8];
    const f16* pb1 = &sB[buf][wn * 32 + 16 + lrow][half * 8];
    v16h a0 = cat8(*(const v8h*)pa0, *(const v8h*)(pa0 + 16));
    v16h a1 = cat8(*(const v8h*)pa1, *(const v8h*)(pa1 + 16));
    v16h b0 = cat8(*(const v8h*)pb0, *(const v8h*)(pb0 + 16));
    v16h b1 = cat8(*(const v8h*)pb1, *(const v8h*)(pb1 + 16));
    acc[0][0] = wmma_f16(a0, b0, acc[0][0]);
    acc[0][1] = wmma_f16(a0, b1, acc[0][1]);
    acc[1][0] = wmma_f16(a1, b0, acc[1][0]);
    acc[1][1] = wmma_f16(a1, b1, acc[1][1]);
    __syncthreads();
  }

  f16* Pb = P + (size_t)b * D * D;
  const int col = lane & 15, hb = lane >> 4;
#pragma unroll
  for (int i = 0; i < 2; ++i) {
#pragma unroll
    for (int j = 0; j < 2; ++j) {
      const int n = nBlk + wn * 32 + j * 16 + col;      // d2
      const int m = mBlk + wm * 32 + i * 16 + hb * 8;   // d1 (8 consecutive)
      v8h pk;
#pragma unroll
      for (int r = 0; r < 8; ++r) pk[r] = (f16)acc[i][j][r];
      *(v8h*)(Pb + (size_t)n * D + m) = pk;  // P[d2][d1]
    }
  }
}

// ---------------------------------------------------------------------------
// Kernel 3 (swapped orientation): C[d2][t] = sum_d1 P[d2][d1] * Q[t][d1].
// A = P rows, B = Q rows; transposed store hits out[t][d2] with 8 consecutive
// f32 per lane (2x b128). grid = (D/128, M/64), block = 256.
// ---------------------------------------------------------------------------
__global__ __launch_bounds__(256) void attn_out_kernel(
    const f16* __restrict__ Qh, const f16* __restrict__ P,
    float* __restrict__ out) {
  constexpr int D = 768;
  const float scale = 0.03608439182435161f;  // 1/sqrt(768)
  const int lane = threadIdx.x & 31;
  const int wave = threadIdx.x >> 5;
  const int wm = wave & 3, wn = wave >> 2;
  const int m0 = blockIdx.x * 128 + wm * 32;      // d2
  const int blockN = blockIdx.y * 64;             // t (global over B*S)
  const int n0 = blockN + wn * 32;
  const int b = blockN >> 12;                     // 64-wide t-tiles never straddle
  const f16* Ap = P + (size_t)b * D * D;          // A = P[d2][d1]

  v8f acc[2][2] = {};
  for (int k0 = 0; k0 < D; k0 += 32) {
    v16h a0 = load_a_f16(Ap, D, m0, k0, lane);
    v16h a1 = load_a_f16(Ap, D, m0 + 16, k0, lane);
    v16h b0 = load_b_f16(Qh, D, n0, k0, lane);      // B[k=d1][n=t] = Q[t][d1]
    v16h b1 = load_b_f16(Qh, D, n0 + 16, k0, lane);
    acc[0][0] = wmma_f16(a0, b0, acc[0][0]);
    acc[0][1] = wmma_f16(a0, b1, acc[0][1]);
    acc[1][0] = wmma_f16(a1, b0, acc[1][0]);
    acc[1][1] = wmma_f16(a1, b1, acc[1][1]);
  }

  const int col = lane & 15, hb = lane >> 4;
#pragma unroll
  for (int i = 0; i < 2; ++i) {
#pragma unroll
    for (int j = 0; j < 2; ++j) {
      const int t = n0 + j * 16 + col;       // global row of out (fixed per lane)
      const int m = m0 + i * 16 + hb * 8;    // d2: 8 consecutive columns
      v4f s0, s1;
#pragma unroll
      for (int r = 0; r < 4; ++r) {
        s0[r] = acc[i][j][r] * scale;
        s1[r] = acc[i][j][4 + r] * scale;
      }
      float* o = out + (size_t)t * D + m;
      *(v4f*)(o) = s0;
      *(v4f*)(o + 4) = s1;
    }
  }
}

// ---------------------------------------------------------------------------
extern "C" void kernel_launch(void* const* d_in, const int* in_sizes, int n_in,
                              void* d_out, int out_size, void* d_ws,
                              size_t ws_size, hipStream_t stream) {
  (void)in_sizes; (void)n_in; (void)out_size; (void)ws_size;
  constexpr int Bn = 4, S = 4096, D = 768, M = Bn * S;

  const float* x  = (const float*)d_in[0];
  const float* Wq = (const float*)d_in[1];
  const float* bq = (const float*)d_in[2];
  const float* Wk = (const float*)d_in[3];
  const float* bk = (const float*)d_in[4];
  const float* Wv = (const float*)d_in[5];
  const float* bv = (const float*)d_in[6];

  // Workspace (f16): xh | Wqh | Wkh | Wvh | Qh | Kt | Vt | P   (~108 MB)
  f16* xh  = (f16*)d_ws;
  f16* Wqh = xh  + (size_t)M * D;
  f16* Wkh = Wqh + (size_t)D * D;
  f16* Wvh = Wkh + (size_t)D * D;
  f16* Qh  = Wvh + (size_t)D * D;
  f16* Kt  = Qh  + (size_t)M * D;
  f16* Vt  = Kt  + (size_t)Bn * D * S;
  f16* P   = Vt  + (size_t)Bn * D * S;

  dim3 blk(256);

  // Kernel 0: one-shot f32 -> f16 conversions (8 elems / thread)
  cvt_f16_kernel<<<dim3((size_t)M * D / 2048), blk, 0, stream>>>(x, xh);
  cvt_f16_kernel<<<dim3((size_t)D * D / 2048), blk, 0, stream>>>(Wq, Wqh);
  cvt_f16_kernel<<<dim3((size_t)D * D / 2048), blk, 0, stream>>>(Wk, Wkh);
  cvt_f16_kernel<<<dim3((size_t)D * D / 2048), blk, 0, stream>>>(Wv, Wvh);

  dim3 g1(M / 128, D / 64);  // 128 x 12
  qkv_proj_kernel<<<g1, blk, 0, stream>>>(xh, Wqh, bq, Wkh, bk, Wvh, bv, Qh, Kt,
                                          Vt);

  dim3 g2(D / 128, D / 64, Bn);  // 6 x 12 x 4
  kv_outer_kernel<<<g2, blk, 0, stream>>>(Kt, Vt, P);

  dim3 g3(D / 128, M / 64);  // 6 x 256
  attn_out_kernel<<<g3, blk, 0, stream>>>(Qh, P, (float*)d_out);
}
